// BidirRecurrentModel_76166950028137
// MI455X (gfx1250) — matrix operand order repaired and checked by reading, not compiled, not run on hardware
//
#include <hip/hip_runtime.h>
#include <hip/hip_bf16.h>

typedef __bf16 bf16;
typedef __attribute__((ext_vector_type(16))) __bf16 v16bf;
typedef __attribute__((ext_vector_type(8)))  __bf16 v8bf;
typedef __attribute__((ext_vector_type(8)))  float   v8f;

#define T_TOT 2048
#define BB 64
#define HH 128
#define GG 512
#define CHUNK 128
#define NCHUNK (T_TOT / CHUNK)

__device__ __forceinline__ float sigf(float x) {
  return 1.0f / (1.0f + __expf(-x));
}
__device__ __forceinline__ float tanh_fast(float x) {
  x = fminf(fmaxf(x, -15.0f), 15.0f);
  float e = __expf(2.0f * x);
  return (e - 1.0f) / (e + 1.0f);
}

__device__ __forceinline__ v8f wmma_bf16(v16bf a, v16bf b, v8f c) {
  // D = A(16x32 bf16) x B(32x16 bf16) + C(16x16 f32)
  return __builtin_amdgcn_wmma_f32_16x16x32_bf16(false, a, false, b, (short)0, c,
                                                 false, false);
}

// B fragment (32x16): B[k][n] = W[n][k], W row-major [N][ldw].
// lanes 0-15: N=lane,  K=k0+{0..7, 16..23} ; lanes 16-31: N=lane-16, K=k0+{8..15, 24..31}
__device__ __forceinline__ v16bf load_b_frag(const bf16* __restrict__ W, int n0,
                                             int k0, int ldw, int lane) {
  int n  = n0 + (lane & 15);
  int kk = k0 + ((lane >> 4) << 3);
  const bf16* p = W + (size_t)n * ldw + kk;
  v8bf lo = *(const v8bf*)(p);
  v8bf hi = *(const v8bf*)(p + 16);
  v16bf b;
#pragma unroll
  for (int i = 0; i < 8; ++i) { b[i] = lo[i]; b[i + 8] = hi[i]; }
  return b;
}

// A fragment (16x32) from bf16 row-major [M][lda] at (r0, k0).
// lanes 0-15 hold M=row, K = k0+{0..7, 16..23}; lanes 16-31: K = k0+{8..15, 24..31}
__device__ __forceinline__ v16bf load_a_frag_bf16(const bf16* __restrict__ A,
                                                  int r0, int k0, int lda, int lane) {
  const bf16* p = A + (size_t)(r0 + (lane & 15)) * lda + k0 + ((lane >> 4) << 3);
  v8bf lo = *(const v8bf*)(p);
  v8bf hi = *(const v8bf*)(p + 16);
  v16bf a;
#pragma unroll
  for (int i = 0; i < 8; ++i) { a[i] = lo[i]; a[i + 8] = hi[i]; }
  return a;
}

// ---------------------------------------------------------------- convert
__global__ void convert_kernel(
    const float* __restrict__ WihF, const float* __restrict__ WhhF,
    const float* __restrict__ bihF, const float* __restrict__ bhhF,
    const float* __restrict__ WihR, const float* __restrict__ WhhR,
    const float* __restrict__ bihR, const float* __restrict__ bhhR,
    const float* __restrict__ fcw,
    bf16* __restrict__ wihbf,   // [L][2][512][128]
    bf16* __restrict__ whhbf,   // [L][2][512][128]
    bf16* __restrict__ fcwbf,   // [128][256]
    float* __restrict__ bsum) { // [L][2][512] = bih + bhh
  int stride = gridDim.x * blockDim.x;
  int tid0 = blockIdx.x * blockDim.x + threadIdx.x;
  for (int idx = tid0; idx < 2 * GG * HH; idx += stride) {
    int l = idx / (GG * HH);
    int r = idx - l * (GG * HH);
    wihbf[(l * 2 + 0) * (GG * HH) + r] = (bf16)WihF[idx];
    wihbf[(l * 2 + 1) * (GG * HH) + r] = (bf16)WihR[idx];
    whhbf[(l * 2 + 0) * (GG * HH) + r] = (bf16)WhhF[idx];
    whhbf[(l * 2 + 1) * (GG * HH) + r] = (bf16)WhhR[idx];
  }
  for (int idx = tid0; idx < 128 * 256; idx += stride)
    fcwbf[idx] = (bf16)fcw[idx];
  for (int idx = tid0; idx < 2 * 2 * GG; idx += stride) {
    int l = idx / (2 * GG);
    int dir = (idx / GG) & 1;
    int g = idx & (GG - 1);
    bsum[idx] = dir ? (bihR[l * GG + g] + bhhR[l * GG + g])
                    : (bihF[l * GG + g] + bhhF[l * GG + g]);
  }
}

// ------------------------------------------------- input projection GEMM
// gx[dir][t][btile][ntile][lane][8] = xs @ Wih^T + (bih+bhh), fragment-native.
// LAYER is a template parameter so the A-source path is straight-line code.
template <int LAYER>
__global__ __launch_bounds__(256) void gemm_kernel(
    const float* __restrict__ x,     // [64][2048][128]  (layer 0 source)
    const bf16*  __restrict__ ysc,   // [2][CHUNK][64][128] (layer 1 source)
    const bf16*  __restrict__ wihl,  // [2][512][128]  (this layer)
    const float* __restrict__ bsuml, // [2][512]
    float* __restrict__ gx,          // [2][CHUNK][4][32][32][8]
    int chunk) {
  int wave = threadIdx.x >> 5, lane = threadIdx.x & 31;
  int task = blockIdx.x * 8 + wave;  // 8192 tasks = 2 dirs * 128 t * 32 ntiles
  int dir  = task >> 12;
  int strip = task & 4095;
  int tl = strip >> 5;   // time within chunk
  int nt = strip & 31;   // gate-column tile
  int rlo = lane & 15;
  int rowadd = (lane >> 4) << 3;

  const bf16* W = wihl + (size_t)dir * GG * HH;
  v16bf Bf[4];
#pragma unroll
  for (int k = 0; k < 4; ++k) Bf[k] = load_b_frag(W, nt * 16, k * 32, HH, lane);
  float bias = bsuml[dir * GG + nt * 16 + rlo];

  int s  = chunk * CHUNK + tl;
  int tg = dir ? (T_TOT - 1 - s) : s;  // reverse direction reads reversed time

#pragma unroll
  for (int mt = 0; mt < 4; ++mt) {     // 4 batch-row tiles (64 rows)
    v8f acc;
#pragma unroll
    for (int r = 0; r < 8; ++r) acc[r] = bias;
    int b = mt * 16 + rlo;
#pragma unroll
    for (int k = 0; k < 4; ++k) {
      v16bf a;
      if (LAYER == 0) {
        const float* px = x + ((size_t)b * T_TOT + tg) * HH + k * 32 + rowadd;
        v8f lo = *(const v8f*)px;
        v8f hi = *(const v8f*)(px + 16);
#pragma unroll
        for (int i = 0; i < 8; ++i) { a[i] = (bf16)lo[i]; a[i + 8] = (bf16)hi[i]; }
      } else {
        const bf16* py = ysc + (((size_t)dir * CHUNK + tl) * BB + b) * HH + k * 32 + rowadd;
        v8bf lo = *(const v8bf*)py;
        v8bf hi = *(const v8bf*)(py + 16);
#pragma unroll
        for (int i = 0; i < 8; ++i) { a[i] = lo[i]; a[i + 8] = hi[i]; }
      }
      acc = wmma_bf16(a, Bf[k], acc);
    }
    // fragment-native store: one contiguous v8f per lane (2x b128)
    float* pg = gx + ((((size_t)dir * CHUNK + tl) * 4 + mt) * 32 + nt) * 256 + lane * 8;
    *(v8f*)pg = acc;
  }
}

// ------------------------------------------------------- persistent scan
// One workgroup per direction; 16 waves; wave = (mtb, jt) block; Whh^T frags
// live in VGPRs; h lives in LDS (bf16); c lives in per-wave registers.
template <int LAYER>
__global__ __launch_bounds__(512) void scan_kernel(
    const float* __restrict__ gx,    // [2][CHUNK][4][32][32][8]
    const bf16*  __restrict__ whhl,  // [2][512][128]   (this layer)
    bf16*  __restrict__ hstl,        // [2][64][128]    (h carry, bf16)
    float* __restrict__ cstl,        // [2][64][128]    (c carry, f32)
    bf16*  __restrict__ ysc,         // [2][CHUNK][64][128] (layer-0 output)
    float* __restrict__ hout,        // [2][64][128]    (layer-1 slices)
    int chunk) {
  int dir  = blockIdx.x;
  int tid  = threadIdx.x;
  int wave = tid >> 5;
  int lane = tid & 31;
  int jt   = wave & 7;               // hidden-unit tile 0..7
  int mtb  = (wave >> 3) << 1;       // batch-row tiles {0,1} or {2,3}
  int rlo  = lane & 15;
  int rowadd = (lane >> 4) << 3;

  __shared__ bf16 hsh[BB * HH];      // 16 KB

  const float* gxd = gx + (size_t)dir * CHUNK * 32768;
  const bf16*  Whh = whhl + (size_t)dir * GG * HH;
  bf16*  hst = hstl + (size_t)dir * BB * HH;
  float* cst = cstl + (size_t)dir * BB * HH;

  // Persistent Whh^T B-fragments: [gate][k] — 128 VGPRs, loop-invariant.
  v16bf Bf[4][4];
#pragma unroll
  for (int g = 0; g < 4; ++g)
#pragma unroll
    for (int k = 0; k < 4; ++k)
      Bf[g][k] = load_b_frag(Whh, g * HH + jt * 16, k * 32, HH, lane);

  for (int i = tid; i < BB * HH / 8; i += 512)
    ((v8bf*)hsh)[i] = ((const v8bf*)hst)[i];

  v8f c[2];
#pragma unroll
  for (int m = 0; m < 2; ++m)
#pragma unroll
    for (int r = 0; r < 8; ++r)
      c[m][r] = cst[((mtb + m) * 16 + rowadd + r) * HH + jt * 16 + rlo];

  __syncthreads();

  for (int t = 0; t < CHUNK; ++t) {
    const float* gxt = gxd + (size_t)t * 32768;
    // acc init = gx tile (fragment-native, contiguous 32B per lane)
    v8f acc[2][4];
#pragma unroll
    for (int m = 0; m < 2; ++m)
#pragma unroll
      for (int g = 0; g < 4; ++g)
        acc[m][g] = *(const v8f*)(gxt + (((mtb + m) * 32 + g * 8 + jt) * 32 + lane) * 8);

    // gates += h @ Whh^T : 32 WMMAs per wave per step
#pragma unroll
    for (int k = 0; k < 4; ++k) {
      v16bf a0 = load_a_frag_bf16(hsh, (mtb + 0) * 16, k * 32, HH, lane);
      v16bf a1 = load_a_frag_bf16(hsh, (mtb + 1) * 16, k * 32, HH, lane);
#pragma unroll
      for (int g = 0; g < 4; ++g) {
        acc[0][g] = wmma_bf16(a0, Bf[g][k], acc[0][g]);
        acc[1][g] = wmma_bf16(a1, Bf[g][k], acc[1][g]);
      }
    }
    if (t + 1 < CHUNK) {  // warm caches for next step's gx
      const float* pf = gxd + (size_t)(t + 1) * 32768;
#pragma unroll
      for (int m = 0; m < 2; ++m)
        __builtin_prefetch(pf + (((mtb + m) * 32 + jt) * 32 + lane) * 8, 0, 3);
    }

    __syncthreads();  // all waves finished reading h
#pragma unroll
    for (int m = 0; m < 2; ++m) {
#pragma unroll
      for (int r = 0; r < 8; ++r) {
        float iv = sigf(acc[m][0][r]);
        float fv = sigf(acc[m][1][r]);
        float gv = tanh_fast(acc[m][2][r]);
        float ov = sigf(acc[m][3][r]);
        float cn = fv * c[m][r] + iv * gv;
        c[m][r] = cn;
        float hv = ov * tanh_fast(cn);
        int row = (mtb + m) * 16 + rowadd + r;
        int col = jt * 16 + rlo;
        bf16 hb = (bf16)hv;
        hsh[row * HH + col] = hb;
        if (LAYER == 0) {
          ysc[((size_t)dir * CHUNK + t) * BB * HH + row * HH + col] = hb;
        } else {
          int s = chunk * CHUNK + t;
          if ((dir == 0 && s == T_TOT - 1) || (dir == 1 && s == 0))
            hout[(size_t)dir * BB * HH + row * HH + col] = hv;
        }
      }
    }
    __syncthreads();  // h fully updated before next step reads it
  }

  for (int i = tid; i < BB * HH / 8; i += 512)
    ((v8bf*)hst)[i] = ((const v8bf*)hsh)[i];
#pragma unroll
  for (int m = 0; m < 2; ++m)
#pragma unroll
    for (int r = 0; r < 8; ++r)
      cst[((mtb + m) * 16 + rowadd + r) * HH + jt * 16 + rlo] = c[m][r];
}

// ------------------------------------------------------------ FC head
__global__ __launch_bounds__(256) void fc_kernel(
    const float* __restrict__ hout,  // [2][64][128]: [xf[-1] | xr[0]]
    const bf16*  __restrict__ fcwbf, // [128][256]
    const float* __restrict__ fcb,
    float* __restrict__ out) {       // [64][128]
  int wave = threadIdx.x >> 5, lane = threadIdx.x & 31;
  int nt = wave;  // 8 waves -> 8 output column tiles
  v16bf Bf[8];
#pragma unroll
  for (int k = 0; k < 8; ++k) Bf[k] = load_b_frag(fcwbf, nt * 16, k * 32, 256, lane);
  float bias = fcb[nt * 16 + (lane & 15)];
  int rowadd = (lane >> 4) << 3;
#pragma unroll
  for (int mt = 0; mt < 4; ++mt) {
    v8f acc;
#pragma unroll
    for (int r = 0; r < 8; ++r) acc[r] = bias;
    int row = mt * 16 + (lane & 15);
#pragma unroll
    for (int k = 0; k < 8; ++k) {
      const float* src = (k < 4) ? hout : (hout + BB * HH);
      int kk = (k < 4) ? k * 32 : (k - 4) * 32;
      const float* p = src + (size_t)row * HH + kk + rowadd;
      v8f lo = *(const v8f*)p;
      v8f hi = *(const v8f*)(p + 16);
      v16bf a;
#pragma unroll
      for (int i = 0; i < 8; ++i) { a[i] = (bf16)lo[i]; a[i + 8] = (bf16)hi[i]; }
      acc = wmma_bf16(a, Bf[k], acc);
    }
    int col = nt * 16 + (lane & 15);
#pragma unroll
    for (int r = 0; r < 8; ++r)
      out[(mt * 16 + rowadd + r) * HH + col] = acc[r];
  }
}

// ---------------------------------------------------------------- launch
extern "C" void kernel_launch(void* const* d_in, const int* in_sizes, int n_in,
                              void* d_out, int out_size, void* d_ws, size_t ws_size,
                              hipStream_t stream) {
  (void)in_sizes; (void)n_in; (void)out_size; (void)ws_size;
  const float* x    = (const float*)d_in[0];
  const float* WihF = (const float*)d_in[1];
  const float* WhhF = (const float*)d_in[2];
  const float* bihF = (const float*)d_in[3];
  const float* bhhF = (const float*)d_in[4];
  const float* WihR = (const float*)d_in[5];
  const float* WhhR = (const float*)d_in[6];
  const float* bihR = (const float*)d_in[7];
  const float* bhhR = (const float*)d_in[8];
  const float* fcw  = (const float*)d_in[9];
  const float* fcb  = (const float*)d_in[10];

  // workspace layout (~38 MB), all 4KB-aligned offsets
  char* ws = (char*)d_ws;
  bf16*  wihbf = (bf16*)(ws + 0);          // 512 KB  [L][2][512][128]
  bf16*  whhbf = (bf16*)(ws + 524288);     // 512 KB
  bf16*  fcwbf = (bf16*)(ws + 1048576);    // 64 KB
  float* bsum  = (float*)(ws + 1114112);   // 8 KB    [L][2][512]
  float* gx    = (float*)(ws + 1122304);   // 32 MB   [2][CHUNK][4][32][32][8]
  bf16*  ysc   = (bf16*)(ws + 34676736);   // 4 MB    [2][CHUNK][64][128]
  bf16*  hst   = (bf16*)(ws + 38871040);   // 128 KB  [L][2][64][128]
  float* cst   = (float*)(ws + 39002112);  // 256 KB  [L][2][64][128]
  float* hout  = (float*)(ws + 39264256);  // 64 KB   [2][64][128]

  hipMemsetAsync(hst, 0, 131072, stream);
  hipMemsetAsync(cst, 0, 262144, stream);

  convert_kernel<<<256, 256, 0, stream>>>(WihF, WhhF, bihF, bhhF, WihR, WhhR,
                                          bihR, bhhR, fcw, wihbf, whhbf, fcwbf, bsum);

  for (int chunk = 0; chunk < NCHUNK; ++chunk) {
    // layer 0
    gemm_kernel<0><<<1024, 256, 0, stream>>>(x, ysc, wihbf, bsum, gx, chunk);
    scan_kernel<0><<<2, 512, 0, stream>>>(gx, whhbf, hst, cst, ysc, hout, chunk);
    // layer 1 (consumes this chunk's ysc)
    gemm_kernel<1><<<1024, 256, 0, stream>>>(x, ysc, wihbf + 2 * GG * HH,
                                             bsum + 2 * GG, gx, chunk);
    scan_kernel<1><<<2, 512, 0, stream>>>(gx, whhbf + 2 * GG * HH,
                                          hst + 2 * BB * HH, cst + 2 * BB * HH,
                                          ysc, hout, chunk);
  }
  fc_kernel<<<1, 256, 0, stream>>>(hout, fcwbf, fcb, (float*)d_out);
}
